// TrajectoryRegressor_30648886624477
// MI455X (gfx1250) — matrix operand order repaired and checked by reading, not compiled, not run on hardware
//
#include <hip/hip_runtime.h>
#include <hip/hip_bf16.h>

typedef __attribute__((ext_vector_type(16))) _Float16 v16h;
typedef __attribute__((ext_vector_type(8)))  _Float16 v8h;
typedef __attribute__((ext_vector_type(8)))  float    v8f;

#define NN   131072
#define EE   2097152
#define SS   512
#define FIN  64
#define HID  128
#define RNNH 256

union V16U { v16h v; v8h h[2]; };

// ---------------- generic fill ----------------
__global__ void fill_f32_kernel(float* __restrict__ p, float v, int n) {
    int i = blockIdx.x * blockDim.x + threadIdx.x;
    if (i < n) p[i] = v;
}

// ---------------- degree accumulation (deg starts at 1.0 for self loop) ----
__global__ void deg_acc_kernel(float* __restrict__ deg, const int* __restrict__ dst,
                               const float* __restrict__ w, int e) {
    int i = blockIdx.x * blockDim.x + threadIdx.x;
    if (i < e) atomicAdd(&deg[dst[i]], w[i]);
}

__global__ void rsqrt_kernel(float* __restrict__ d, int n) {
    int i = blockIdx.x * blockDim.x + threadIdx.x;
    if (i < n) {
        float v = d[i];
        d[i] = (v > 0.0f) ? rsqrtf(fmaxf(v, 1e-12f)) : 0.0f;
    }
}

// ---------------- weight fp32 -> f16 transposed [Nout][K] -------------------
__global__ void wt_convert_kernel(const float* __restrict__ W, _Float16* __restrict__ Wt,
                                  int K, int Nout) {
    int i = blockIdx.x * blockDim.x + threadIdx.x;
    if (i < K * Nout) {
        int k = i / Nout, n = i % Nout;
        Wt[n * K + k] = (_Float16)W[i];
    }
}

// ---------------- WMMA GEMM: C[M x 128] = A[M x K] * W[K x 128] -------------
// One workgroup (256 thr / 8 waves) computes a 16-row strip; wave w owns cols
// [16w, 16w+16). A strip staged in LDS as f16; B read from pre-transposed f16
// weights (two 16B runs per lane fragment). f32 accumulate via
// v_wmma_f32_16x16x32_f16.
__global__ void __launch_bounds__(256)
gcn_gemm_kernel(const float* __restrict__ A, const _Float16* __restrict__ Wt,
                float* __restrict__ C, int K) {
    __shared__ _Float16 As[16 * 128];
    const int m0 = blockIdx.x * 16;
    // stage A strip (row-major, stride K) with fp32->f16 convert
    for (int i = threadIdx.x; i < 16 * K; i += 256)
        As[i] = (_Float16)A[(size_t)m0 * K + i];
    __syncthreads();

    const int wave    = threadIdx.x >> 5;
    const int lane    = threadIdx.x & 31;
    const int lm      = lane & 15;
    const int halfSel = lane >> 4;          // 0: lanes 0-15, 1: lanes 16-31
    const int n0      = wave * 16;
    const int col     = n0 + lm;

    v8f c = {};
    for (int k0 = 0; k0 < K; k0 += 32) {
        // A frag: row lm, halves {kb..kb+7, kb+16..kb+23}, kb = 8*halfSel
        const _Float16* ar = &As[lm * K + k0 + 8 * halfSel];
        V16U a; a.h[0] = *(const v8h*)(ar); a.h[1] = *(const v8h*)(ar + 16);
        // B frag: col, halves {kb..kb+7, kb+16..kb+23}, kb = 8*halfSel
        const _Float16* br = Wt + (size_t)col * K + k0 + 8 * halfSel;
        V16U b; b.h[0] = *(const v8h*)(br); b.h[1] = *(const v8h*)(br + 16);
        c = __builtin_amdgcn_wmma_f32_16x16x32_f16(false, a.v, false, b.v,
                                                   (short)0, c, false, false);
    }
    // f32 C/D layout: vgpr v -> row m0 + 8*halfSel + v, column = col
    const int rbase = m0 + 8 * halfSel;
#pragma unroll
    for (int v = 0; v < 8; v++)
        C[(size_t)(rbase + v) * 128 + col] = c[v];
}

// ---------------- edge aggregation: F[dst] += norm_e * G[src] ---------------
// one wave per edge, each lane moves 4 consecutive floats (float4 gather,
// 4x global_atomic_add_f32 scatter)
__global__ void __launch_bounds__(256)
agg_kernel(const float* __restrict__ G, const float* __restrict__ dinv,
           const int* __restrict__ src, const int* __restrict__ dst,
           const float* __restrict__ w, float* __restrict__ F, int e) {
    long long gid = (long long)blockIdx.x * blockDim.x + threadIdx.x;
    int ei = (int)(gid >> 5);
    if (ei >= e) return;
    int lane = (int)(gid & 31);
    int s = src[ei], d = dst[ei];
    float nrm = dinv[s] * w[ei] * dinv[d];
    int f = lane * 4;
    float4 g = *(const float4*)(G + (size_t)s * 128 + f);
    float* o = F + (size_t)d * 128 + f;
    atomicAdd(o + 0, nrm * g.x);
    atomicAdd(o + 1, nrm * g.y);
    atomicAdd(o + 2, nrm * g.z);
    atomicAdd(o + 3, nrm * g.w);
}

// ---------------- epilogue: F = relu(F + dinv^2 * G + b) --------------------
__global__ void post_kernel(float* __restrict__ F, const float* __restrict__ G,
                            const float* __restrict__ dinv, const float* __restrict__ b,
                            int n) {
    int i = blockIdx.x * blockDim.x + threadIdx.x;
    if (i < n * 128) {
        int node = i >> 7, f = i & 127;
        float di = dinv[node];
        float v = F[i] + di * di * G[i] + b[f];
        F[i] = v > 0.0f ? v : 0.0f;
    }
}

// ---------------- mean pool accumulation ------------------------------------
__global__ void __launch_bounds__(256)
pool_acc_kernel(const float* __restrict__ Z, const int* __restrict__ batch,
                float* __restrict__ psum, float* __restrict__ pcnt, int n) {
    long long gid = (long long)blockIdx.x * blockDim.x + threadIdx.x;
    int node = (int)(gid >> 5);
    if (node >= n) return;
    int lane = (int)(gid & 31);
    int s = batch[node];
    int f = lane * 4;
    float4 z = *(const float4*)(Z + (size_t)node * 128 + f);
    float* o = psum + (size_t)s * 128 + f;
    atomicAdd(o + 0, z.x);
    atomicAdd(o + 1, z.y);
    atomicAdd(o + 2, z.z);
    atomicAdd(o + 3, z.w);
    if (lane == 0) atomicAdd(&pcnt[s], 1.0f);
}

__global__ void pool_div_kernel(const float* __restrict__ psum, const float* __restrict__ pcnt,
                                float* __restrict__ pooled) {
    int i = blockIdx.x * blockDim.x + threadIdx.x;
    if (i < SS * 128) {
        int s = i >> 7;
        pooled[i] = psum[i] / fmaxf(pcnt[s], 1.0f);
    }
}

// ---------------- serial tanh RNN + sigmoid head ----------------------------
__global__ void __launch_bounds__(256)
rnn_kernel(const float* __restrict__ pooled, const float* __restrict__ Wih,
           const float* __restrict__ Whh, const float* __restrict__ bih,
           const float* __restrict__ bhh, const float* __restrict__ Wl,
           const float* __restrict__ bl, float* __restrict__ out) {
    __shared__ float h[RNNH];
    __shared__ float xt[HID];
    const int i = threadIdx.x;
    h[i] = 0.0f;
    const float bi = bih[i] + bhh[i];
    const float* wi = Wih + (size_t)i * HID;
    const float* wh = Whh + (size_t)i * RNNH;
    __syncthreads();
    for (int t = 0; t < SS; t++) {
        if (i < HID) xt[i] = pooled[(size_t)t * HID + i];
        __syncthreads();
        float acc = bi;
#pragma unroll 8
        for (int j = 0; j < HID; j++)  acc += wi[j] * xt[j];
#pragma unroll 8
        for (int k = 0; k < RNNH; k++) acc += wh[k] * h[k];
        float hv = tanhf(acc);
        __syncthreads();
        h[i] = hv;
        __syncthreads();
        if (i < 2) {
            float l = bl[i];
            const float* wl = Wl + (size_t)i * RNNH;
#pragma unroll 8
            for (int k = 0; k < RNNH; k++) l += wl[k] * h[k];
            out[t * 2 + i] = 1.0f / (1.0f + expf(-l));
        }
        __syncthreads();
    }
}

extern "C" void kernel_launch(void* const* d_in, const int* in_sizes, int n_in,
                              void* d_out, int out_size, void* d_ws, size_t ws_size,
                              hipStream_t stream) {
    const float* x    = (const float*)d_in[0];
    const float* ew   = (const float*)d_in[1];
    const int*   src  = (const int*)d_in[2];
    const int*   dst  = (const int*)d_in[3];
    const int*   batch= (const int*)d_in[4];
    const float* W1   = (const float*)d_in[5];
    const float* b1   = (const float*)d_in[6];
    const float* W2   = (const float*)d_in[7];
    const float* b2   = (const float*)d_in[8];
    const float* W3   = (const float*)d_in[9];
    const float* b3   = (const float*)d_in[10];
    const float* Wih  = (const float*)d_in[11];
    const float* Whh  = (const float*)d_in[12];
    const float* bih  = (const float*)d_in[13];
    const float* bhh  = (const float*)d_in[14];
    const float* Wl   = (const float*)d_in[15];
    const float* bl   = (const float*)d_in[16];
    float* out = (float*)d_out;

    char* wsb = (char*)d_ws;
    size_t off = 0;
    auto take = [&](size_t bytes) -> char* {
        char* p = wsb + off;
        off = (off + bytes + 255) & ~(size_t)255;
        return p;
    };
    float*     dinv   = (float*)take((size_t)NN * 4);
    float*     G      = (float*)take((size_t)NN * 128 * 4);
    float*     F      = (float*)take((size_t)NN * 128 * 4);
    _Float16*  W1t    = (_Float16*)take((size_t)FIN * 128 * 2);
    _Float16*  W2t    = (_Float16*)take((size_t)128 * 128 * 2);
    _Float16*  W3t    = (_Float16*)take((size_t)128 * 128 * 2);
    float*     psum   = (float*)take((size_t)SS * 128 * 4);
    float*     pcnt   = (float*)take((size_t)SS * 4);
    float*     pooled = (float*)take((size_t)SS * 128 * 4);

    const int T = 256;
    auto blocks = [](long long n, int t) { return (unsigned)((n + t - 1) / t); };

    // degrees -> dinv (in place)
    fill_f32_kernel<<<blocks(NN, T), T, 0, stream>>>(dinv, 1.0f, NN);
    deg_acc_kernel<<<blocks(EE, T), T, 0, stream>>>(dinv, dst, ew, EE);
    rsqrt_kernel<<<blocks(NN, T), T, 0, stream>>>(dinv, NN);

    // f16 transposed weights
    wt_convert_kernel<<<blocks(FIN * 128, T), T, 0, stream>>>(W1, W1t, FIN, 128);
    wt_convert_kernel<<<blocks(128 * 128, T), T, 0, stream>>>(W2, W2t, 128, 128);
    wt_convert_kernel<<<blocks(128 * 128, T), T, 0, stream>>>(W3, W3t, 128, 128);

    const float* layer_in[3] = { x, F, F };
    const _Float16* layer_w[3] = { W1t, W2t, W3t };
    const float* layer_b[3] = { b1, b2, b3 };
    const int layer_k[3] = { FIN, 128, 128 };

    for (int L = 0; L < 3; L++) {
        gcn_gemm_kernel<<<NN / 16, 256, 0, stream>>>(layer_in[L], layer_w[L], G, layer_k[L]);
        fill_f32_kernel<<<blocks((long long)NN * 128, T), T, 0, stream>>>(F, 0.0f, NN * 128);
        agg_kernel<<<blocks((long long)EE * 32, T), T, 0, stream>>>(G, dinv, src, dst, ew, F, EE);
        post_kernel<<<blocks((long long)NN * 128, T), T, 0, stream>>>(F, G, dinv, layer_b[L], NN);
    }

    // mean pool per subgraph
    fill_f32_kernel<<<blocks(SS * 128, T), T, 0, stream>>>(psum, 0.0f, SS * 128);
    fill_f32_kernel<<<blocks(SS, T), T, 0, stream>>>(pcnt, 0.0f, SS);
    pool_acc_kernel<<<blocks((long long)NN * 32, T), T, 0, stream>>>(F, batch, psum, pcnt, NN);
    pool_div_kernel<<<blocks(SS * 128, T), T, 0, stream>>>(psum, pcnt, pooled);

    // serial RNN + sigmoid head
    rnn_kernel<<<1, 256, 0, stream>>>(pooled, Wih, Whh, bih, bhh, Wl, bl, out);
}
